// PhaseSimilarity_7138235646543
// MI455X (gfx1250) — matrix operand order repaired and checked by reading, not compiled, not run on hardware
//
#include <hip/hip_runtime.h>

typedef __attribute__((ext_vector_type(16))) _Float16 v16h;
typedef __attribute__((ext_vector_type(8)))  float    v8f;
typedef __attribute__((ext_vector_type(4)))  float    v4f;

#define PS_B 2
#define PS_H 8
#define PS_T 2048
#define PS_A 64
#define PS_N 64

// One workgroup per (b,t): 8 waves, wave w handles head h=w (H==8).
// Each wave computes 64 outputs (a=0..63) as 4 tiles of 16 rows via
// v_wmma_f32_16x16x32_f16: A = cos/sin of k-angles (16 rows x 32 K),
// B = [cq | sq | 0...] (32 K x 16 N, 2 useful columns), f32 accumulate.
__global__ __launch_bounds__(256) void PhaseSimilarity_wmma_kernel(
    const float* __restrict__ qang,          // (B,T,N)
    const float* __restrict__ kang,          // (B,H,T,A,N)
    const unsigned char* __restrict__ valid, // (B,H,T,A) bool
    float* __restrict__ out)                 // (B,H,T,A) f32
{
  __shared__ float s_cq[PS_N];
  __shared__ float s_sq[PS_N];
  __shared__ float s_scr[8][2][16][16]; // per-wave D-fragment scratch (16 KB)

  const int bt = blockIdx.x;          // 0 .. B*T-1
  const int b  = bt >> 11;            // T == 2048
  const int t  = bt & (PS_T - 1);

  const int tid  = threadIdx.x;
  const int wave = tid >> 5;          // wave32
  const int lane = tid & 31;
  const int half = lane >> 4;
  const int l    = lane & 15;

  // Stage cos/sin of q into LDS (shared by all 8 heads of this (b,t)).
  if (tid < PS_N) {
    float ang = qang[((size_t)b * PS_T + t) * PS_N + tid];
    s_cq[tid] = __cosf(ang);
    s_sq[tid] = __sinf(ang);
  }
  __syncthreads();

  // Build B fragments once (identical for every tile / head of this block).
  // 16-bit B-matrix 32x16 layout: lane holds column n=l; elements 0..15 map
  // to K = s*32 + half*8 + (i<8 ? i : i+8). Column 0 = cq, column 1 = sq, rest zero.
  v16h bfrag[2];
#pragma unroll
  for (int s = 0; s < 2; ++s) {
#pragma unroll
    for (int i = 0; i < 16; ++i) {
      const int kk = s * 32 + half * 8 + (i < 8 ? i : i + 8);
      const float v = (l == 0) ? s_cq[kk] : ((l == 1) ? s_sq[kk] : 0.0f);
      bfrag[s][i] = (_Float16)v;
    }
  }

  const int h = wave;
  const float* kbase =
      kang + (((size_t)b * PS_H + h) * PS_T + t) * (size_t)(PS_A * PS_N);

#pragma unroll
  for (int tile = 0; tile < 4; ++tile) {
    const int a0 = tile * 16;
    const float* kr = kbase + (size_t)(a0 + l) * PS_N; // row M = l

    v8f accC = {0.f,0.f,0.f,0.f,0.f,0.f,0.f,0.f}; // cos(k) matrix accum
    v8f accS = {0.f,0.f,0.f,0.f,0.f,0.f,0.f,0.f}; // sin(k) matrix accum

#pragma unroll
    for (int s = 0; s < 2; ++s) {
      // A-matrix 16x32 f16 layout: lane<16 holds K = {0..7, 16..23}+32s,
      // lane>=16 holds K = {8..15, 24..31}+32s. Streamed, non-temporal.
      const int base = s * 32 + half * 8;
      v4f d0 = __builtin_nontemporal_load((const v4f*)(kr + base));
      v4f d1 = __builtin_nontemporal_load((const v4f*)(kr + base + 4));
      v4f d2 = __builtin_nontemporal_load((const v4f*)(kr + base + 16));
      v4f d3 = __builtin_nontemporal_load((const v4f*)(kr + base + 20));

      float w[16];
#pragma unroll
      for (int i = 0; i < 4; ++i) {
        w[i] = d0[i]; w[4 + i] = d1[i]; w[8 + i] = d2[i]; w[12 + i] = d3[i];
      }

      // Transcendentals on the VALU pipe; MACs go to the matrix pipe.
      v16h ac, as2;
#pragma unroll
      for (int i = 0; i < 16; ++i) {
        ac[i]  = (_Float16)__cosf(w[i]);
        as2[i] = (_Float16)__sinf(w[i]);
      }

      accC = __builtin_amdgcn_wmma_f32_16x16x32_f16(
          false, ac,  false, bfrag[s], (short)0, accC, false, false);
      accS = __builtin_amdgcn_wmma_f32_16x16x32_f16(
          false, as2, false, bfrag[s], (short)0, accS, false, false);
    }

    // D layout: element j, lane<16 -> (M=j, N=l); lane>=16 -> (M=j+8, N=l).
    // Only columns 0,1 are meaningful; round-trip through per-wave LDS so 16
    // lanes can each finish one row.
#pragma unroll
    for (int j = 0; j < 8; ++j) {
      const int m = j + half * 8;
      s_scr[wave][0][m][l] = accC[j];
      s_scr[wave][1][m][l] = accS[j];
    }
    // DS ops are in-order within a wave; just drain the store counter.
    asm volatile("s_wait_dscnt 0" ::: "memory");

    if (half == 0) {
      // row m = l:  R1=cq.ck  I1=sq.ck  I2=cq.sk  R2=sq.sk
      const float re = s_scr[wave][0][l][0] + s_scr[wave][1][l][1];
      const float im = s_scr[wave][0][l][1] - s_scr[wave][1][l][0];
      const float mag = __builtin_sqrtf(re * re + im * im) * (1.0f / PS_N);
      const size_t oi =
          (((size_t)b * PS_H + h) * PS_T + t) * (size_t)PS_A + (a0 + l);
      out[oi] = valid[oi] ? mag : -__builtin_inff();
    }
    // keep next tile's LDS stores after this tile's LDS reads
    asm volatile("" ::: "memory");
  }
}

extern "C" void kernel_launch(void* const* d_in, const int* in_sizes, int n_in,
                              void* d_out, int out_size, void* d_ws, size_t ws_size,
                              hipStream_t stream) {
  const float*         qang  = (const float*)d_in[0];
  const float*         kang  = (const float*)d_in[1];
  const unsigned char* valid = (const unsigned char*)d_in[2];
  // d_in[3] = batch, d_in[4] = time (scalars; shapes are compile-time here)
  float* out = (float*)d_out;

  dim3 grid(PS_B * PS_T); // 4096 workgroups, one per (b,t)
  dim3 block(256);        // 8 wave32 waves = 8 heads
  PhaseSimilarity_wmma_kernel<<<grid, block, 0, stream>>>(qang, kang, valid, out);
}
